// AttentionSubsample_51470888075485
// MI455X (gfx1250) — matrix operand order repaired and checked by reading, not compiled, not run on hardware
//
#include <hip/hip_runtime.h>

// ---------------------------------------------------------------------------
// AttentionSubsample fused pipeline for gfx1250 (MI455X), wave32 + WMMA bf16.
// All matmuls use v_wmma_f32_16x16x32_bf16. BN affine + 1/sqrt(KD) folded
// into bf16 weights at prep time. Flash-style online softmax per 16-row tile.
// ---------------------------------------------------------------------------

typedef __attribute__((ext_vector_type(16))) __bf16 v16bf;
typedef __attribute__((ext_vector_type(8)))  float  v8f;

#define Bz   32
#define Nn   1280
#define N_q  320
#define Cc   256
#define Hh   8
#define KDd  64
#define Dd   128
#define DHh  1024
#define HKV  1536
#define NHKD 512
#define OUTC 384
#define EPSf 1e-5f

__device__ __forceinline__ v8f wmma_bf16(v16bf a, v16bf b, v8f c) {
  return __builtin_amdgcn_wmma_f32_16x16x32_bf16(false, a, false, b, (short)0, c,
                                                 false, false);
}

// A fragment (16x32 bf16, MxK) from a per-lane row pointer already offset to
// rowBase + k0 + (laneHalf*8). Pairs 0..3 hold K=half*8..+7, pairs 4..7 hold
// K=16+half*8..+7 (ISA 7.12.2, 16-bit A 16x32).
__device__ __forceinline__ v16bf a_from_rowptr(const __bf16* __restrict__ p) {
  v16bf a;
#pragma unroll
  for (int j = 0; j < 8; ++j) a[j] = p[j];
#pragma unroll
  for (int j = 0; j < 8; ++j) a[8 + j] = p[16 + j];
  return a;
}

__device__ __forceinline__ v16bf load_a_frag(const __bf16* __restrict__ base,
                                             long row0, int ld, int k0) {
  int lane = threadIdx.x & 31;
  int half = lane >> 4;
  int m    = lane & 15;
  return a_from_rowptr(base + (row0 + m) * (long)ld + k0 + half * 8);
}

// B fragment (32x16 bf16, KxN) sourced from a row-major [col][K] array
// (i.e. B^T). Lane = column, elems 0..7 K=k0+half*8..+7, 8..15 K=k0+16+half*8..+7.
__device__ __forceinline__ v16bf load_bt_frag(const __bf16* __restrict__ base,
                                              long col0, int ld, int k0) {
  int lane = threadIdx.x & 31;
  int half = lane >> 4;
  int n    = lane & 15;
  const __bf16* p = base + (col0 + n) * (long)ld + k0 + half * 8;
  v16bf b;
#pragma unroll
  for (int j = 0; j < 8; ++j) { b[j] = p[j]; b[8 + j] = p[16 + j]; }
  return b;
}

// ---------------------------------------------------------------------------
// Prep kernels
// ---------------------------------------------------------------------------

__global__ void k_cvt_bf16(const float* __restrict__ in, __bf16* __restrict__ out,
                           long n) {
  long i = (long)blockIdx.x * blockDim.x + threadIdx.x;
  long st = (long)gridDim.x * blockDim.x;
  for (; i < n; i += st) out[i] = (__bf16)in[i];
}

// Fold BN: y = (xW^T - rm)*s + b, s = g*rsqrt(rv+eps)  =>  W' = W*s, b' = b-rm*s.
// kv_mode: also fold attention scale 1/8 into k-channels (ch%192 < 64).
__global__ void k_prep_w(const float* __restrict__ W, const float* __restrict__ g,
                         const float* __restrict__ b, const float* __restrict__ rm,
                         const float* __restrict__ rv, __bf16* __restrict__ Wout,
                         float* __restrict__ bout, int rows, int cols, int kv_mode) {
  long n = (long)rows * cols;
  long i = (long)blockIdx.x * blockDim.x + threadIdx.x;
  long st = (long)gridDim.x * blockDim.x;
  for (; i < n; i += st) {
    int r = (int)(i / cols);
    int c = (int)(i - (long)r * cols);
    float s = g[r] * rsqrtf(rv[r] + EPSf);
    float extra = 1.0f;
    if (kv_mode && (r % 192) < KDd) extra = 0.125f;  // KD^-0.5 = 1/8
    Wout[i] = (__bf16)(W[i] * s * extra);
    if (c == 0) bout[r] = (b[r] - rm[r] * s) * extra;
  }
}

// biasmat[h][q][k] = attn_bias[h][bias_idxs[q][k]]  (13 MB, L2-resident)
__global__ void k_prep_bias(const float* __restrict__ attn_bias,
                            const int* __restrict__ idxs,
                            float* __restrict__ biasmat, int n_off) {
  long n = (long)Hh * N_q * Nn;
  long i = (long)blockIdx.x * blockDim.x + threadIdx.x;
  long st = (long)gridDim.x * blockDim.x;
  for (; i < n; i += st) {
    int h = (int)(i / ((long)N_q * Nn));
    long r = i - (long)h * N_q * Nn;  // q*Nn + k
    biasmat[i] = attn_bias[(long)h * n_off + idxs[r]];
  }
}

// ---------------------------------------------------------------------------
// GEMM 1: kv = x(40960x256) * Wkv^T(256x1536) + bias  -> K bf16, V^T bf16
// Wave computes 16x64; 24 col-groups x 2560 row-tiles = 61440 waves.
// ---------------------------------------------------------------------------
__global__ void __launch_bounds__(256) k_gemm_kv(
    const __bf16* __restrict__ x, const __bf16* __restrict__ W,
    const float* __restrict__ bias, __bf16* __restrict__ Kmat,
    __bf16* __restrict__ Vt) {
  int wave = blockIdx.x * (blockDim.x >> 5) + (threadIdx.x >> 5);
  const int NG = HKV / 64;  // 24
  int mt = wave / NG;
  int ng = wave - mt * NG;
  if (mt >= (Bz * Nn) / 16) return;
  long row0 = (long)mt * 16;
  int col0 = ng * 64;
  int lane = threadIdx.x & 31, half = lane >> 4, lc = lane & 15;

  v8f acc[4] = {};
  for (int kk = 0; kk < Cc; kk += 32) {
    v16bf a = load_a_frag(x, row0, Cc, kk);
#pragma unroll
    for (int t = 0; t < 4; ++t) {
      v16bf bf = load_bt_frag(W, col0 + t * 16, Cc, kk);
      acc[t] = wmma_bf16(a, bf, acc[t]);
    }
  }
#pragma unroll
  for (int t = 0; t < 4; ++t) {
    int j = col0 + t * 16 + lc;
    int h = j / 192;
    int jj = j - h * 192;
    float bj = bias[j];
#pragma unroll
    for (int r = 0; r < 8; ++r) {
      long row = row0 + half * 8 + r;
      long bb = row / Nn;
      long n = row - bb * Nn;
      float v = acc[t][r] + bj;
      if (jj < KDd) {  // K: [b][h][token][kd]
        Kmat[(((bb * Hh) + h) * Nn + n) * KDd + jj] = (__bf16)v;
      } else {         // V^T: [b][h][d][token]
        Vt[(((bb * Hh) + h) * Dd + (jj - KDd)) * Nn + n] = (__bf16)v;
      }
    }
  }
}

// ---------------------------------------------------------------------------
// GEMM 2: q = subsample(x)(10240x256) * Wq^T(256x512) + bias -> Q bf16
// ---------------------------------------------------------------------------
__device__ __forceinline__ int subsrc(int t) {
  if (t < 256) { int ty = t >> 4, tx = t & 15; return ty * 64 + tx * 2; }
  int u = t - 256; int ty = u >> 3, tx = u & 7;
  return 1024 + ty * 32 + tx * 2;
}

__global__ void __launch_bounds__(256) k_gemm_q(
    const __bf16* __restrict__ x, const __bf16* __restrict__ W,
    const float* __restrict__ bias, __bf16* __restrict__ Qmat) {
  int wave = blockIdx.x * (blockDim.x >> 5) + (threadIdx.x >> 5);
  const int NG = NHKD / 64;  // 8
  int mt = wave / NG;
  int ng = wave - mt * NG;
  if (mt >= (Bz * N_q) / 16) return;
  int lane = threadIdx.x & 31, half = lane >> 4, lc = lane & 15;
  int row0 = mt * 16;
  int b = row0 / N_q;
  int t0 = row0 - b * N_q;  // tiles never cross batch (320 % 16 == 0)
  const __bf16* arow = x + ((long)b * Nn + subsrc(t0 + lc)) * Cc;
  int col0 = ng * 64;

  v8f acc[4] = {};
  for (int kk = 0; kk < Cc; kk += 32) {
    v16bf a = a_from_rowptr(arow + kk + half * 8);
#pragma unroll
    for (int t = 0; t < 4; ++t) {
      v16bf bf = load_bt_frag(W, col0 + t * 16, Cc, kk);
      acc[t] = wmma_bf16(a, bf, acc[t]);
    }
  }
#pragma unroll
  for (int t = 0; t < 4; ++t) {
    int j = col0 + t * 16 + lc;
    int h = j >> 6, kd = j & 63;
    float bj = bias[j];
#pragma unroll
    for (int r = 0; r < 8; ++r) {
      int tq = t0 + half * 8 + r;
      Qmat[(((long)b * Hh + h) * N_q + tq) * KDd + kd] = (__bf16)(acc[t][r] + bj);
    }
  }
}

// ---------------------------------------------------------------------------
// Flash attention: one wave per (b,h,q-tile-of-16). Online softmax over
// N=1280 keys in chunks of 64. P transposed through LDS into A-fragments.
// Output fused with hard_swish, stored bf16 [b][q][h*128+d].
// ---------------------------------------------------------------------------
__global__ void __launch_bounds__(256) k_attn(
    const __bf16* __restrict__ Qmat, const __bf16* __restrict__ Kmat,
    const __bf16* __restrict__ Vt, const float* __restrict__ biasmat,
    __bf16* __restrict__ Omat) {
  __shared__ __bf16 Pbuf[8][16 * 64];
  int widx = threadIdx.x >> 5;
  int wave = blockIdx.x * 8 + widx;
  int b = wave / (Hh * 20);
  int rem = wave - b * (Hh * 20);
  int h = rem / 20;
  int qt = rem - h * 20;
  int lane = threadIdx.x & 31, half = lane >> 4, lc = lane & 15;

  const __bf16* Qb = Qmat + (((long)b * Hh + h) * N_q + qt * 16) * KDd;
  const __bf16* Kb = Kmat + ((long)b * Hh + h) * Nn * KDd;
  const __bf16* Vb = Vt + ((long)b * Hh + h) * Dd * Nn;
  const float* Bb = biasmat + ((long)h * N_q + qt * 16) * Nn;

  v16bf q0 = load_a_frag(Qb, 0, KDd, 0);
  v16bf q1 = load_a_frag(Qb, 0, KDd, 32);

  v8f o[8] = {};
  float mrow[8], lrow[8];
#pragma unroll
  for (int r = 0; r < 8; ++r) { mrow[r] = -1e30f; lrow[r] = 0.0f; }

  for (int kc = 0; kc < Nn; kc += 64) {
    // S = Q*K^T (+ bias, scale already folded into K)
    v8f s[4] = {};
#pragma unroll
    for (int t = 0; t < 4; ++t) {
      s[t] = wmma_bf16(q0, load_bt_frag(Kb, kc + t * 16, KDd, 0), s[t]);
      s[t] = wmma_bf16(q1, load_bt_frag(Kb, kc + t * 16, KDd, 32), s[t]);
    }
    float mc[8];
#pragma unroll
    for (int r = 0; r < 8; ++r) {
      const float* bp = Bb + (long)(half * 8 + r) * Nn + kc + lc;
      float mx = -1e30f;
#pragma unroll
      for (int t = 0; t < 4; ++t) {
        float v = s[t][r] + bp[t * 16];
        s[t][r] = v;
        mx = fmaxf(mx, v);
      }
      mc[r] = mx;
    }
#pragma unroll
    for (int off = 1; off < 16; off <<= 1) {
#pragma unroll
      for (int r = 0; r < 8; ++r) mc[r] = fmaxf(mc[r], __shfl_xor(mc[r], off, 32));
    }
    float alpha[8], rs[8];
#pragma unroll
    for (int r = 0; r < 8; ++r) {
      float mn = fmaxf(mrow[r], mc[r]);
      alpha[r] = __expf(mrow[r] - mn);
      mrow[r] = mn;
      float sum = 0.0f;
#pragma unroll
      for (int t = 0; t < 4; ++t) {
        float p = __expf(s[t][r] - mn);
        s[t][r] = p;
        sum += p;
      }
      rs[r] = sum;
    }
#pragma unroll
    for (int off = 1; off < 16; off <<= 1) {
#pragma unroll
      for (int r = 0; r < 8; ++r) rs[r] += __shfl_xor(rs[r], off, 32);
    }
#pragma unroll
    for (int r = 0; r < 8; ++r) lrow[r] = lrow[r] * alpha[r] + rs[r];
#pragma unroll
    for (int d = 0; d < 8; ++d)
#pragma unroll
      for (int r = 0; r < 8; ++r) o[d][r] *= alpha[r];

    // Transpose P (C-layout) -> A-fragment layout via LDS
#pragma unroll
    for (int t = 0; t < 4; ++t)
#pragma unroll
      for (int r = 0; r < 8; ++r)
        Pbuf[widx][(half * 8 + r) * 64 + t * 16 + lc] = (__bf16)s[t][r];
    __syncthreads();
    v16bf p0 = load_a_frag(&Pbuf[widx][0], 0, 64, 0);
    v16bf p1 = load_a_frag(&Pbuf[widx][0], 0, 64, 32);
    __syncthreads();

    // O += P*V  (V^T stored [d][token], contiguous per lane)
#pragma unroll
    for (int d = 0; d < 8; ++d) {
      o[d] = wmma_bf16(p0, load_bt_frag(Vb, d * 16, Nn, kc), o[d]);
      o[d] = wmma_bf16(p1, load_bt_frag(Vb, d * 16, Nn, kc + 32), o[d]);
    }
  }

  float inv[8];
#pragma unroll
  for (int r = 0; r < 8; ++r) inv[r] = 1.0f / lrow[r];
  long orow0 = (long)b * N_q + qt * 16;
#pragma unroll
  for (int d = 0; d < 8; ++d) {
#pragma unroll
    for (int r = 0; r < 8; ++r) {
      float v = o[d][r] * inv[r];
      float hs = v * fminf(fmaxf(v + 3.0f, 0.0f), 6.0f) * (1.0f / 6.0f);
      Omat[(orow0 + half * 8 + r) * DHh + h * Dd + d * 16 + lc] = (__bf16)hs;
    }
  }
}

// ---------------------------------------------------------------------------
// GEMM 3: out = O(10240x1024) * Wp^T(1024x384) + bias  (f32 output)
// ---------------------------------------------------------------------------
__global__ void __launch_bounds__(256) k_gemm_p(
    const __bf16* __restrict__ Omat, const __bf16* __restrict__ W,
    const float* __restrict__ bias, float* __restrict__ out) {
  int wave = blockIdx.x * (blockDim.x >> 5) + (threadIdx.x >> 5);
  const int NG = OUTC / 64;  // 6
  int mt = wave / NG;
  int ng = wave - mt * NG;
  if (mt >= (Bz * N_q) / 16) return;
  long row0 = (long)mt * 16;
  int col0 = ng * 64;
  int lane = threadIdx.x & 31, half = lane >> 4, lc = lane & 15;

  v8f acc[4] = {};
  for (int kk = 0; kk < DHh; kk += 32) {
    v16bf a = load_a_frag(Omat, row0, DHh, kk);
#pragma unroll
    for (int t = 0; t < 4; ++t) {
      v16bf bf = load_bt_frag(W, col0 + t * 16, DHh, kk);
      acc[t] = wmma_bf16(a, bf, acc[t]);
    }
  }
#pragma unroll
  for (int t = 0; t < 4; ++t) {
    int j = col0 + t * 16 + lc;
    float bj = bias[j];
#pragma unroll
    for (int r = 0; r < 8; ++r)
      out[(row0 + half * 8 + r) * OUTC + j] = acc[t][r] + bj;
  }
}

// ---------------------------------------------------------------------------

extern "C" void kernel_launch(void* const* d_in, const int* in_sizes, int n_in,
                              void* d_out, int out_size, void* d_ws, size_t ws_size,
                              hipStream_t stream) {
  const float* x     = (const float*)d_in[0];
  const float* W_kv  = (const float*)d_in[1];
  const float* g_kv  = (const float*)d_in[2];
  const float* b_kv  = (const float*)d_in[3];
  const float* rm_kv = (const float*)d_in[4];
  const float* rv_kv = (const float*)d_in[5];
  const float* W_q   = (const float*)d_in[6];
  const float* g_q   = (const float*)d_in[7];
  const float* b_q   = (const float*)d_in[8];
  const float* rm_q  = (const float*)d_in[9];
  const float* rv_q  = (const float*)d_in[10];
  const float* W_p   = (const float*)d_in[11];
  const float* g_p   = (const float*)d_in[12];
  const float* b_p   = (const float*)d_in[13];
  const float* rm_p  = (const float*)d_in[14];
  const float* rv_p  = (const float*)d_in[15];
  const float* attn_bias = (const float*)d_in[16];
  const int*   bias_idxs = (const int*)d_in[17];
  int n_off = in_sizes[16] / Hh;

  char* ws = (char*)d_ws;
  size_t off = 0;
  auto alloc = [&](size_t bytes) -> void* {
    void* p = ws + off;
    off += (bytes + 255) & ~(size_t)255;
    return p;
  };
  __bf16* x_bf   = (__bf16*)alloc((size_t)Bz * Nn * Cc * 2);
  __bf16* Wkv_bf = (__bf16*)alloc((size_t)HKV * Cc * 2);
  float*  bkv    = (float*)alloc((size_t)HKV * 4);
  __bf16* Wq_bf  = (__bf16*)alloc((size_t)NHKD * Cc * 2);
  float*  bq     = (float*)alloc((size_t)NHKD * 4);
  __bf16* Wp_bf  = (__bf16*)alloc((size_t)OUTC * DHh * 2);
  float*  bp     = (float*)alloc((size_t)OUTC * 4);
  __bf16* Kmat   = (__bf16*)alloc((size_t)Bz * Hh * Nn * KDd * 2);
  __bf16* Vt     = (__bf16*)alloc((size_t)Bz * Hh * Dd * Nn * 2);
  __bf16* Qmat   = (__bf16*)alloc((size_t)Bz * Hh * N_q * KDd * 2);
  float*  biasm  = (float*)alloc((size_t)Hh * N_q * Nn * 4);
  __bf16* Omat   = (__bf16*)alloc((size_t)Bz * N_q * DHh * 2);
  (void)ws_size;

  // --- prep ---
  k_cvt_bf16<<<4096, 256, 0, stream>>>(x, x_bf, (long)Bz * Nn * Cc);
  k_prep_w<<<1536, 256, 0, stream>>>(W_kv, g_kv, b_kv, rm_kv, rv_kv,
                                     Wkv_bf, bkv, HKV, Cc, 1);
  k_prep_w<<<512, 256, 0, stream>>>(W_q, g_q, b_q, rm_q, rv_q,
                                    Wq_bf, bq, NHKD, Cc, 0);
  k_prep_w<<<1536, 256, 0, stream>>>(W_p, g_p, b_p, rm_p, rv_p,
                                     Wp_bf, bp, OUTC, DHh, 0);
  k_prep_bias<<<4096, 256, 0, stream>>>(attn_bias, bias_idxs, biasm, n_off);

  // --- main pipeline ---
  k_gemm_kv<<<7680, 256, 0, stream>>>(x_bf, Wkv_bf, bkv, Kmat, Vt);
  k_gemm_q<<<640, 256, 0, stream>>>(x_bf, Wq_bf, bq, Qmat);
  k_attn<<<640, 256, 0, stream>>>(Qmat, Kmat, Vt, biasm, Omat);
  k_gemm_p<<<480, 256, 0, stream>>>(Omat, Wp_bf, bp, (float*)d_out);
}
